// CausalSelfAttention_57861799412149
// MI455X (gfx1250) — compile-verified, matrix operand order checked by reading, not run on hardware
//
#include <hip/hip_runtime.h>

// ---------------------------------------------------------------------------
// CDNA5 (gfx1250) fused attention block.
// Wave32. Matmuls via v_wmma_f32_16x16x32_bf16; LDS staging via the CDNA5
// async global->LDS path (global_load_async_to_lds_b128 + s_wait_asynccnt),
// double-buffered.
// ---------------------------------------------------------------------------

typedef __attribute__((ext_vector_type(16))) __bf16 v16bf;
typedef __attribute__((ext_vector_type(8)))  float  v8f;

#define B_SZ   2
#define T_SEQ  2048
#define N_EMBD 1024
#define NH     16
#define NKV    4
#define HD     64
#define QKV_N  1536   // 1024 q + 256 k + 256 v

union V16U { v16bf v; uint4 q[2]; __bf16 s[16]; };

static __device__ inline v8f wmma_bf16(v16bf a, v16bf b, v8f c) {
  // D = A(16x32 bf16) x B(32x16 bf16) + C(16x16 f32)
  return __builtin_amdgcn_wmma_f32_16x16x32_bf16(false, a, false, b, (short)0, c,
                                                 false, false);
}

// CDNA5 async global->LDS copy, 16B per lane. VDST = LDS byte offset
// (low 32 bits of the flat shared-aperture address, ISA 10.2). ASYNCcnt.
static __device__ inline void async_copy_b128(const void* gptr, void* lptr) {
  unsigned lds_off = (unsigned)(unsigned long long)lptr;
  asm volatile("global_load_async_to_lds_b128 %0, %1, off"
               :: "v"(lds_off), "v"(gptr) : "memory");
}
static __device__ inline void wait_async0() {
  asm volatile("s_wait_asynccnt 0" ::: "memory");
}

// A/B fragment loader: "row" matrix stored [16+ rows][pitch] bf16.
// Lane l holds row (l&15); k-halves per ISA 16-bit A layout:
//   lanes 0-15:  elems 0..7 -> k=0..7,   elems 8..15 -> k=16..23
//   lanes 16-31: elems 0..7 -> k=8..15,  elems 8..15 -> k=24..31
static __device__ inline v16bf load_frag16x32(const __bf16* tile, int pitch) {
  int lane = threadIdx.x & 31;
  const __bf16* p = tile + (size_t)(lane & 15) * pitch + ((lane >> 4) << 3);
  V16U u;
  u.q[0] = *(const uint4*)(p);
  u.q[1] = *(const uint4*)(p + 16);
  return u.v;
}

// B fragment where the WMMA-k dimension is the *row* index of the tile
// (used for P x V: B[k=kvrow][n=dcol]). Column-strided LDS reads.
static __device__ inline v16bf load_frag_col16x32(const __bf16* tile, int pitch,
                                                  int col) {
  int lane = threadIdx.x & 31;
  int c  = col + (lane & 15);
  int kb = (lane >> 4) << 3;
  V16U u;
#pragma unroll
  for (int i = 0; i < 8; ++i) u.s[i]     = tile[(size_t)(kb + i) * pitch + c];
#pragma unroll
  for (int i = 0; i < 8; ++i) u.s[8 + i] = tile[(size_t)(kb + 16 + i) * pitch + c];
  return u.v;
}

// ---------------------------------------------------------------------------
// f32 -> bf16 conversion
// ---------------------------------------------------------------------------
__global__ void cvt_bf16_kernel(const float* __restrict__ src,
                                __bf16* __restrict__ dst, int n) {
  int i = blockIdx.x * blockDim.x + threadIdx.x;
  if (i < n) dst[i] = (__bf16)src[i];
}

// ---------------------------------------------------------------------------
// C[M,N] (f32) = X[M,K](bf16) * W[N,K](bf16)^T
// Block tile 128x64, 8 waves, each wave: 16 rows x 4 n-tiles. K panel = 32.
// Double-buffered LDS fed by async global->LDS copies.
// ---------------------------------------------------------------------------
__global__ __launch_bounds__(256)
void gemm_xwT_kernel(const __bf16* __restrict__ X, const __bf16* __restrict__ W,
                     float* __restrict__ C, int M, int N, int K) {
  __shared__ __attribute__((aligned(16))) __bf16 atb[2][128 * 32];
  __shared__ __attribute__((aligned(16))) __bf16 btb[2][64 * 32];
  const int tid  = threadIdx.x;
  const int wave = tid >> 5;
  const int lane = tid & 31;
  const int m0 = blockIdx.x * 128;
  const int n0 = blockIdx.y * 64;

  auto issue_ab = [&](int k0, int buf) {
    // A tile: 128 rows x 32 cols = 512 x 16B chunks; 2 per thread (async)
#pragma unroll
    for (int c = tid; c < 512; c += 256) {
      int r = c >> 2, ch = c & 3;
      async_copy_b128(X + (size_t)(m0 + r) * K + k0 + ch * 8,
                      &atb[buf][r * 32 + ch * 8]);
    }
    // B tile: 64 rows x 32 cols = 256 chunks; 1 per thread (async)
    {
      int r = tid >> 2, ch = tid & 3;
      async_copy_b128(W + (size_t)(n0 + r) * K + k0 + ch * 8,
                      &btb[buf][r * 32 + ch * 8]);
    }
  };

  v8f acc[4] = {};
  issue_ab(0, 0);
  int cur = 0;

  for (int k0 = 0; k0 < K; k0 += 32) {
    wait_async0();      // my async fills of buf[cur] complete
    __syncthreads();    // everyone's fills complete; prev buf free for reuse
    if (k0 + 32 < K) {
      issue_ab(k0 + 32, cur ^ 1);
      __builtin_prefetch(X + (size_t)(m0 + (tid >> 1)) * K + k0 + 64, 0, 1);
    }
    v16bf af = load_frag16x32(&atb[cur][(wave * 16) * 32], 32);
#pragma unroll
    for (int nt = 0; nt < 4; ++nt) {
      v16bf bfr = load_frag16x32(&btb[cur][(nt * 16) * 32], 32);
      acc[nt] = wmma_bf16(af, bfr, acc[nt]);
    }
    cur ^= 1;
  }

  const int lh = lane >> 4, ln = lane & 15;
#pragma unroll
  for (int nt = 0; nt < 4; ++nt)
#pragma unroll
    for (int v = 0; v < 8; ++v) {
      int row = m0 + wave * 16 + v + 8 * lh;
      int col = n0 + nt * 16 + ln;
      C[(size_t)row * N + col] = acc[nt][v];
    }
}

// ---------------------------------------------------------------------------
// Fused gate + ve add + RoPE + RMSNorm. One wave per head-row.
// Rows per (b,t): 16 q heads, 4 k heads, 4 v heads (24 waves).
// Lane = rope-pair index d in [0,32).
// ---------------------------------------------------------------------------
__global__ __launch_bounds__(256)
void fuse_rope_norm_kernel(const float* __restrict__ qkv,
                           const float* __restrict__ x,
                           const float* __restrict__ ve,
                           const float* __restrict__ cosb,
                           const float* __restrict__ sinb,
                           const float* __restrict__ wgate,
                           __bf16* __restrict__ qo, __bf16* __restrict__ ko,
                           __bf16* __restrict__ vo) {
  const int wid  = blockIdx.x * 8 + (threadIdx.x >> 5);
  const int lane = threadIdx.x & 31;
  const int bt = wid / 24;
  const int r  = wid % 24;
  const int b = bt / T_SEQ, t = bt % T_SEQ;
  const float eps = 1.1920929e-07f;  // finfo(f32).eps

  if (r < 20) {  // q or k head: RoPE + RMSNorm
    const int isQ = (r < 16);
    const int h = isQ ? r : (r - 16);
    const float* src = qkv + (size_t)bt * QKV_N + (isQ ? h * HD : 1024 + h * HD);
    float x1 = src[lane], x2 = src[lane + 32];
    float c = cosb[t * 32 + lane], s = sinb[t * 32 + lane];
    float y1 = x1 * c + x2 * s;
    float y2 = -x1 * s + x2 * c;
    float ss = y1 * y1 + y2 * y2;
#pragma unroll
    for (int m = 1; m < 32; m <<= 1) ss += __shfl_xor(ss, m, 32);
    float sc = rsqrtf(ss / 64.0f + eps);
    y1 *= sc; y2 *= sc;
    __bf16* dst = isQ ? (qo + (((size_t)(b * NH + h)) * T_SEQ + t) * HD)
                      : (ko + (((size_t)(b * NKV + h)) * T_SEQ + t) * HD);
    dst[lane]      = (__bf16)y1;
    dst[lane + 32] = (__bf16)y2;
  } else {  // v head: gate + ve
    const int kh = r - 20;
    float gp = x[(size_t)bt * N_EMBD + lane] * wgate[kh * 32 + lane];
#pragma unroll
    for (int m = 1; m < 32; m <<= 1) gp += __shfl_xor(gp, m, 32);
    float g = 2.0f / (1.0f + __expf(-gp));
    const float* src   = qkv + (size_t)bt * QKV_N + 1280 + kh * HD;
    const float* vesrc = ve + (size_t)bt * (NKV * HD) + kh * HD;
    float v1 = src[lane]      + g * vesrc[lane];
    float v2 = src[lane + 32] + g * vesrc[lane + 32];
    __bf16* dst = vo + (((size_t)(b * NKV + kh)) * T_SEQ + t) * HD;
    dst[lane]      = (__bf16)v1;
    dst[lane + 32] = (__bf16)v2;
  }
}

// ---------------------------------------------------------------------------
// Flash attention (flipped-causal window: valid iff qi <= kj <= qi + window).
// 64 queries / block (4 waves x 16 rows), KV blocks of 32, double-buffered
// async K/V staging. S (16x32 f32) -> masked online softmax -> bf16 P via
// per-wave LDS -> O += P x V (4 WMMAs over head-dim tiles).
// ---------------------------------------------------------------------------
#define ATT_BQ  64
#define ATT_BKV 32

__global__ __launch_bounds__(128)
void attn_kernel(const __bf16* __restrict__ qg, const __bf16* __restrict__ kg,
                 const __bf16* __restrict__ vg, __bf16* __restrict__ y,
                 const int* __restrict__ win_ptr) {
  __shared__ __attribute__((aligned(16))) __bf16 ktb[2][ATT_BKV * HD];
  __shared__ __attribute__((aligned(16))) __bf16 vtb[2][ATT_BKV * HD];
  __shared__ __attribute__((aligned(16))) __bf16 pt[4][16 * ATT_BKV];

  const int tid  = threadIdx.x;
  const int wave = tid >> 5;
  const int lane = tid & 31;
  const int lh = lane >> 4, ln = lane & 15;

  const int nqb = T_SEQ / ATT_BQ;  // 32
  const int qb = blockIdx.x % nqb;
  const int h  = (blockIdx.x / nqb) % NH;
  const int b  = blockIdx.x / (nqb * NH);
  const int kvh = h >> 2;  // GQA: 4 q heads per kv head

  const int window = win_ptr[0];
  const float scale = 0.125f;  // 64^-0.5
  const float NEG = -1e30f;

  const __bf16* Qb = qg + ((size_t)(b * NH + h))    * T_SEQ * HD;
  const __bf16* Kb = kg + ((size_t)(b * NKV + kvh)) * T_SEQ * HD;
  const __bf16* Vb = vg + ((size_t)(b * NKV + kvh)) * T_SEQ * HD;

  const int qs = qb * ATT_BQ;
  const int qrow0 = qs + wave * 16;

  auto issue_kv = [&](int blk, int buf) {
    const int kv0 = qs + blk * ATT_BKV;
    // 32 rows x 8x16B chunks = 256 chunks / 128 threads (async to LDS)
#pragma unroll
    for (int c = tid; c < 256; c += 128) {
      int r = c >> 3, ch = c & 7;
      int sr = kv0 + r; if (sr > T_SEQ - 1) sr = T_SEQ - 1;
      async_copy_b128(Kb + (size_t)sr * HD + ch * 8, &ktb[buf][r * HD + ch * 8]);
      async_copy_b128(Vb + (size_t)sr * HD + ch * 8, &vtb[buf][r * HD + ch * 8]);
    }
  };

  // Q A-fragments for both 32-wide head-dim panels (held in VGPRs all loop)
  v16bf qf0 = load_frag16x32(Qb + (size_t)qrow0 * HD + 0,  HD);
  v16bf qf1 = load_frag16x32(Qb + (size_t)qrow0 * HD + 32, HD);

  v8f o0 = {}, o1 = {}, o2 = {}, o3 = {};
  float run_max[8], run_sum[8];
#pragma unroll
  for (int i = 0; i < 8; ++i) { run_max[i] = NEG; run_sum[i] = 0.f; }

  int kv_last = qs + ATT_BQ - 1 + window;
  if (kv_last > T_SEQ - 1) kv_last = T_SEQ - 1;
  const int nkv = (kv_last - qs + ATT_BKV) / ATT_BKV;

  issue_kv(0, 0);
  int cur = 0;

  for (int blk = 0; blk < nkv; ++blk) {
    const int kv0 = qs + blk * ATT_BKV;
    wait_async0();
    __syncthreads();
    if (blk + 1 < nkv) issue_kv(blk + 1, cur ^ 1);
    const __bf16* kt = ktb[cur];
    const __bf16* vt = vtb[cur];

    // S = Q K^T : two 16x16 n-tiles, two K=32 panels each
    v8f s0 = {}, s1 = {};
    {
      v16bf kb0 = load_frag16x32(kt + 0,              HD);
      v16bf kb1 = load_frag16x32(kt + 16 * HD,        HD);
      s0 = wmma_bf16(qf0, kb0, s0);
      s1 = wmma_bf16(qf0, kb1, s1);
      kb0 = load_frag16x32(kt + 32,           HD);
      kb1 = load_frag16x32(kt + 16 * HD + 32, HD);
      s0 = wmma_bf16(qf1, kb0, s0);
      s1 = wmma_bf16(qf1, kb1, s1);
    }

    // masked online softmax; P -> per-wave LDS tile (bf16, 16x32)
#pragma unroll
    for (int v = 0; v < 8; ++v) {
      const int qi  = qrow0 + v + 8 * lh;
      const int kj0 = kv0 + ln;
      const int kj1 = kv0 + 16 + ln;
      float a0 = (kj0 >= qi && kj0 <= qi + window && kj0 < T_SEQ) ? s0[v] * scale : NEG;
      float a1 = (kj1 >= qi && kj1 <= qi + window && kj1 < T_SEQ) ? s1[v] * scale : NEG;
      float mx = fmaxf(a0, a1);
#pragma unroll
      for (int m = 1; m < 16; m <<= 1) mx = fmaxf(mx, __shfl_xor(mx, m, 32));
      float nm   = fmaxf(run_max[v], mx);
      float corr = __expf(run_max[v] - nm);
      float p0 = (a0 > NEG * 0.5f) ? __expf(a0 - nm) : 0.f;
      float p1 = (a1 > NEG * 0.5f) ? __expf(a1 - nm) : 0.f;
      float ps = p0 + p1;
#pragma unroll
      for (int m = 1; m < 16; m <<= 1) ps += __shfl_xor(ps, m, 32);
      run_sum[v] = run_sum[v] * corr + ps;
      run_max[v] = nm;
      o0[v] *= corr; o1[v] *= corr; o2[v] *= corr; o3[v] *= corr;
      const int ml = v + 8 * lh;
      pt[wave][ml * ATT_BKV + ln]      = (__bf16)p0;
      pt[wave][ml * ATT_BKV + 16 + ln] = (__bf16)p1;
    }
    // same-wave LDS producer/consumer: in-order, no barrier needed
    v16bf pf = load_frag16x32(&pt[wave][0], ATT_BKV);
    o0 = wmma_bf16(pf, load_frag_col16x32(vt, HD, 0),  o0);
    o1 = wmma_bf16(pf, load_frag_col16x32(vt, HD, 16), o1);
    o2 = wmma_bf16(pf, load_frag_col16x32(vt, HD, 32), o2);
    o3 = wmma_bf16(pf, load_frag_col16x32(vt, HD, 48), o3);
    cur ^= 1;
  }

  // epilogue: normalize and scatter to y[b, t, h*64 + d] (bf16)
#pragma unroll
  for (int v = 0; v < 8; ++v) {
    float inv = 1.0f / run_sum[v];  // diagonal kj==qi always valid -> sum > 0
    int t = qrow0 + v + 8 * lh;
    size_t base = ((size_t)b * T_SEQ + t) * N_EMBD + h * HD;
    y[base + 0 + ln]  = (__bf16)(o0[v] * inv);
    y[base + 16 + ln] = (__bf16)(o1[v] * inv);
    y[base + 32 + ln] = (__bf16)(o2[v] * inv);
    y[base + 48 + ln] = (__bf16)(o3[v] * inv);
  }
}

// ---------------------------------------------------------------------------
// Host launcher
// ---------------------------------------------------------------------------
extern "C" void kernel_launch(void* const* d_in, const int* in_sizes, int n_in,
                              void* d_out, int out_size, void* d_ws,
                              size_t ws_size, hipStream_t stream) {
  const float* x     = (const float*)d_in[0];
  const float* ve    = (const float*)d_in[1];
  const float* cosb  = (const float*)d_in[2];
  const float* sinb  = (const float*)d_in[3];
  const float* wq    = (const float*)d_in[4];
  const float* wk    = (const float*)d_in[5];
  const float* wv    = (const float*)d_in[6];
  const float* wproj = (const float*)d_in[7];
  const float* wgate = (const float*)d_in[8];
  const int*   win   = (const int*)d_in[9];
  float* out = (float*)d_out;

  const size_t BT = (size_t)B_SZ * T_SEQ;  // 4096

  size_t off = 0;
  auto carve = [&](size_t bytes) {
    char* p = (char*)d_ws + off;
    off += (bytes + 255) & ~(size_t)255;
    return p;
  };
  __bf16* xbf  = (__bf16*)carve(BT * N_EMBD * 2);            // x in bf16
  __bf16* wcat = (__bf16*)carve((size_t)QKV_N * N_EMBD * 2); // [wq;wk;wv] bf16
  __bf16* wpb  = (__bf16*)carve((size_t)N_EMBD * N_EMBD * 2);
  float*  qkv  = (float*)carve(BT * QKV_N * 4);
  __bf16* qbh  = (__bf16*)carve((size_t)B_SZ * NH  * T_SEQ * HD * 2);
  __bf16* kbh  = (__bf16*)carve((size_t)B_SZ * NKV * T_SEQ * HD * 2);
  __bf16* vbh  = (__bf16*)carve((size_t)B_SZ * NKV * T_SEQ * HD * 2);
  __bf16* ybf  = (__bf16*)carve(BT * N_EMBD * 2);
  (void)ws_size; (void)in_sizes; (void)n_in; (void)out_size;

  auto cvt = [&](const float* s, __bf16* d, size_t n) {
    cvt_bf16_kernel<<<(unsigned)((n + 255) / 256), 256, 0, stream>>>(s, d, (int)n);
  };
  cvt(x, xbf, BT * N_EMBD);
  cvt(wq, wcat,                         (size_t)1024 * N_EMBD);
  cvt(wk, wcat + (size_t)1024 * N_EMBD, (size_t)256 * N_EMBD);
  cvt(wv, wcat + (size_t)1280 * N_EMBD, (size_t)256 * N_EMBD);
  cvt(wproj, wpb, (size_t)N_EMBD * N_EMBD);

  // QKV projection: (4096 x 1024) x (1536 x 1024)^T
  gemm_xwT_kernel<<<dim3(32, 24), 256, 0, stream>>>(xbf, wcat, qkv,
                                                    (int)BT, QKV_N, N_EMBD);
  // gate + ve + RoPE + RMSNorm
  fuse_rope_norm_kernel<<<(unsigned)(BT * 24 / 8), 256, 0, stream>>>(
      qkv, x, ve, cosb, sinb, wgate, qbh, kbh, vbh);
  // flash attention
  attn_kernel<<<(unsigned)(B_SZ * NH * (T_SEQ / ATT_BQ)), 128, 0, stream>>>(
      qbh, kbh, vbh, ybf, win);
  // output projection: (4096 x 1024) x (1024 x 1024)^T -> f32 out
  gemm_xwT_kernel<<<dim3(32, 16), 256, 0, stream>>>(ybf, wpb, out,
                                                    (int)BT, N_EMBD, N_EMBD);
}
